// SpatialSemanticAttention_12799002542144
// MI455X (gfx1250) — matrix operand order repaired and checked by reading, not compiled, not run on hardware
//
#include <hip/hip_runtime.h>
#include <hip/hip_bf16.h>

typedef __attribute__((ext_vector_type(16))) __bf16 v16bf;
typedef __attribute__((ext_vector_type(8)))  __bf16 v8bf;
typedef __attribute__((ext_vector_type(8)))  float  v8f;

#define C_DIM 512
#define HW    4096
#define BSTR  (512 * 4096)
#define INV_SQRT_D 0.03608439182435161f  /* 1/sqrt(768) */

__device__ __forceinline__ unsigned short f2bf(float f) {
    unsigned u = __float_as_uint(f);
    u += 0x7FFFu + ((u >> 16) & 1u);   // round-to-nearest-even
    return (unsigned short)(u >> 16);
}

// Load a 16-half B fragment from LDS with only 16B alignment guaranteed.
__device__ __forceinline__ v16bf ld_bfrag(const unsigned short* p) {
    v8bf lo = *(const v8bf*)p;
    v8bf hi = *(const v8bf*)(p + 16);
    return __builtin_shufflevector(lo, hi, 0,1,2,3,4,5,6,7,8,9,10,11,12,13,14,15);
}

// ---------------------------------------------------------------------------
// Precompute 1: Q = (se*w)·wq^T + wq_b ; A_s = (Q·wk_w)/sqrt(768) swizzled to
// WMMA A-fragment layout (rows 8..15 zero-padded); c0 = Q·wk_b/sqrt(768).
// One block per batch.
// ---------------------------------------------------------------------------
__global__ __launch_bounds__(256) void precompute_qa(
    const float* __restrict__ se, const float* __restrict__ wts,
    const float* __restrict__ wq_w, const float* __restrict__ wq_b,
    const float* __restrict__ wk_w, const float* __restrict__ wk_b,
    unsigned short* __restrict__ afr, float* __restrict__ c0)
{
    __shared__ float we[8 * 768];
    __shared__ float Qb[8 * 512];
    const int b = blockIdx.x, tid = threadIdx.x;

    for (int i = tid; i < 8 * 768; i += 256) {
        int n = i / 768;
        we[i] = se[b * 6144 + i] * wts[n];
    }
    __syncthreads();
    for (int i = tid; i < 4096; i += 256) {
        int n = i >> 9, c = i & 511;
        const float* wr = we + n * 768;
        const float* qr = wq_w + c * 768;
        float acc = wq_b[c];
        for (int d = 0; d < 768; ++d) acc += wr[d] * qr[d];
        Qb[i] = acc;
    }
    __syncthreads();
    if (tid < 8) {
        const float* q = Qb + tid * 512;
        float acc = 0.f;
        for (int o = 0; o < 512; ++o) acc += q[o] * wk_b[o];
        c0[b * 8 + tid] = acc * INV_SQRT_D;
    }
    // zero-fill padded rows 8..15 (lanes 8..15 and 24..31 of every fragment)
    for (int i = tid; i < 4096; i += 256) {
        int k = i >> 8, rest = i & 255, lsel = rest >> 4, e = rest & 15;
        int l = (lsel < 8) ? (8 + lsel) : (16 + lsel);
        afr[(((b * 16 + k) * 32 + l) << 4) + e] = 0;
    }
    // scatter A_s rows 0..7 into fragment layout
    for (int i = tid; i < 4096; i += 256) {
        int n = i >> 9, kc = i & 511;
        const float* q = Qb + n * 512;
        float acc = 0.f;
        for (int o = 0; o < 512; ++o) acc += q[o] * wk_w[o * 512 + kc];
        int k = kc >> 5, k0 = kc & 31;
        int l = n + 16 * ((k0 >> 3) & 1);
        int e = (k0 & 7) + 8 * (k0 >> 4);
        afr[(((b * 16 + k) * 32 + l) << 4) + e] = f2bf(acc * INV_SQRT_D);
    }
}

// ---------------------------------------------------------------------------
// Precompute 2: W2 = wo_w·wv_w (bf16, swizzled A-fragment layout), b2 = wo_w·wv_b.
// One block per output channel o.
// ---------------------------------------------------------------------------
__global__ __launch_bounds__(256) void precompute_w2(
    const float* __restrict__ wo_w, const float* __restrict__ wv_w,
    const float* __restrict__ wv_b, unsigned short* __restrict__ w2fr,
    float* __restrict__ b2)
{
    __shared__ float wo_row[512];
    const int o = blockIdx.x, tid = threadIdx.x;
    for (int i = tid; i < 512; i += 256) wo_row[i] = wo_w[o * 512 + i];
    __syncthreads();
    if (tid == 0) {
        float a = 0.f;
        for (int m = 0; m < 512; ++m) a += wo_row[m] * wv_b[m];
        b2[o] = a;
    }
    const int mtile = o >> 4, rr = o & 15;
    for (int c = tid; c < 512; c += 256) {
        float acc = 0.f;
        for (int m = 0; m < 512; ++m) acc += wo_row[m] * wv_w[m * 512 + c];
        int k = c >> 5, k0 = c & 31;
        int l = rr + 16 * ((k0 >> 3) & 1);
        int e = (k0 & 7) + 8 * (k0 >> 4);
        w2fr[(((mtile * 16 + k) * 32 + l) << 4) + e] = f2bf(acc);
    }
}

// ---------------------------------------------------------------------------
// Main fused kernel. Grid (64 rows, 16 batches), 512 threads = 16 waves.
// Each WG: one image row (64 positions). K streamed in 4 chunks of 128 chans.
// Wave w accumulates M-tiles {2w,2w+1} x N-tiles 0..3; waves 0..3 also
// accumulate the 16x64 scores tile.
// ---------------------------------------------------------------------------
__global__ __launch_bounds__(512) void fused_main(
    const float* __restrict__ x,
    const __bf16* __restrict__ w2fr,
    const __bf16* __restrict__ afr,
    const float* __restrict__ c0,
    const float* __restrict__ b2,
    const float* __restrict__ wob,
    float* __restrict__ out)
{
    __shared__ unsigned short xs[64 * 136];   // [pos][chan-in-chunk], padded stride
    __shared__ float scoresLDS[8 * 64];
    __shared__ float mrow[8];
    __shared__ float dinv[8];
    __shared__ float sLDS[64];

    const int tid  = threadIdx.x;
    const int lane = tid & 31;
    const int w    = tid >> 5;        // wave 0..15
    const int b    = blockIdx.y;
    const int h064 = blockIdx.x * 64; // row offset in hw plane
    const long xbase = (long)b * BSTR + h064;

    const int laneq = lane & 15;
    const int lhi   = lane >> 4;      // 0|1
    const int m0    = w * 2;

    v8f zf = {};
    v8f acc[2][4];
#pragma unroll
    for (int t = 0; t < 2; ++t)
#pragma unroll
        for (int nt = 0; nt < 4; ++nt) acc[t][nt] = zf;
    v8f accS = zf;

    for (int kc = 0; kc < 4; ++kc) {
        __syncthreads();
        const int cbase = kc * 128;
#pragma unroll
        for (int it = 0; it < 16; ++it) {
            int i  = tid + it * 512;
            int ch = i >> 6;          // 0..127
            int p  = i & 63;
            float v = x[xbase + (long)(cbase + ch) * HW + p];
            xs[p * 136 + ch] = f2bf(v);
        }
        __syncthreads();
#pragma unroll
        for (int ks = 0; ks < 4; ++ks) {
            const int kg = kc * 4 + ks;   // absolute K-step 0..15
            v16bf a0 = *(const v16bf*)(w2fr + ((((m0 + 0) * 16 + kg) * 32 + lane) << 4));
            v16bf a1 = *(const v16bf*)(w2fr + ((((m0 + 1) * 16 + kg) * 32 + lane) << 4));
            const int krow = ks * 32 + lhi * 8;
            v16bf bb[4];
#pragma unroll
            for (int nt = 0; nt < 4; ++nt)
                bb[nt] = ld_bfrag(xs + (nt * 16 + laneq) * 136 + krow);
#pragma unroll
            for (int nt = 0; nt < 4; ++nt) {
                acc[0][nt] = __builtin_amdgcn_wmma_f32_16x16x32_bf16(
                    false, a0, false, bb[nt], (short)0, acc[0][nt], false, false);
                acc[1][nt] = __builtin_amdgcn_wmma_f32_16x16x32_bf16(
                    false, a1, false, bb[nt], (short)0, acc[1][nt], false, false);
            }
            if (w < 4) {  // wave-uniform: EXEC stays all-ones inside
                v16bf aS = *(const v16bf*)(afr + (((b * 16 + kg) * 32 + lane) << 4));
                v16bf bS = ld_bfrag(xs + (w * 16 + laneq) * 136 + krow);
                accS = __builtin_amdgcn_wmma_f32_16x16x32_bf16(
                    false, aS, false, bS, (short)0, accS, false, false);
            }
        }
    }

    // scores tile -> LDS (rows 0..7 hold n; rows 8..15 are zero padding)
    if (w < 4 && lane < 16) {
        int p = w * 16 + lane;
#pragma unroll
        for (int j = 0; j < 8; ++j)
            scoresLDS[j * 64 + p] = accS[j] + c0[b * 8 + j];
    }
    __syncthreads();

    // softmax over width per n, then s[pos] = sum_n attn
    if (tid < 8) {
        const float* sc = scoresLDS + tid * 64;
        float m = -3.0e38f;
        for (int i = 0; i < 64; ++i) m = fmaxf(m, sc[i]);
        float s = 0.f;
        for (int i = 0; i < 64; ++i) s += __expf(sc[i] - m);
        mrow[tid] = m;
        dinv[tid] = 1.0f / s;
    }
    __syncthreads();
    if (tid < 64) {
        float s = 0.f;
#pragma unroll
        for (int n = 0; n < 8; ++n)
            s += __expf(scoresLDS[n * 64 + tid] - mrow[n]) * dinv[n];
        sLDS[tid] = s;
    }
    __syncthreads();

    // epilogue: out = x + s * (W2 x + b2) + wo_b
#pragma unroll
    for (int t = 0; t < 2; ++t) {
        const int rowb = (m0 + t) * 16 + lhi * 8;
#pragma unroll
        for (int nt = 0; nt < 4; ++nt) {
            const int p = nt * 16 + laneq;
            const float sp = sLDS[p];
#pragma unroll
            for (int j = 0; j < 8; ++j) {
                const int cch = rowb + j;
                const long idx = xbase + (long)cch * HW + p;
                out[idx] = x[idx] + sp * (acc[t][nt][j] + b2[cch]) + wob[cch];
            }
        }
    }
}

// ---------------------------------------------------------------------------
// Workspace layout (bytes):
//   [0,        262144)  A_s fragments, bf16: 16 b * 16 k * 32 lanes * 16 halfs
//   [262144,   262656)  c0: 16*8 f32
//   [262656,   264704)  b2: 512 f32 (+pad)
//   [264704,   788992)  W2 fragments, bf16: 32 m * 16 k * 32 lanes * 16 halfs
// ---------------------------------------------------------------------------
extern "C" void kernel_launch(void* const* d_in, const int* in_sizes, int n_in,
                              void* d_out, int out_size, void* d_ws, size_t ws_size,
                              hipStream_t stream) {
    const float* x    = (const float*)d_in[0];
    const float* se   = (const float*)d_in[1];
    const float* wts  = (const float*)d_in[2];
    const float* wq_w = (const float*)d_in[3];
    const float* wq_b = (const float*)d_in[4];
    const float* wk_w = (const float*)d_in[5];
    const float* wk_b = (const float*)d_in[6];
    const float* wv_w = (const float*)d_in[7];
    const float* wv_b = (const float*)d_in[8];
    const float* wo_w = (const float*)d_in[9];
    const float* wo_b = (const float*)d_in[10];
    float* out = (float*)d_out;

    char* ws = (char*)d_ws;
    unsigned short* afr_w  = (unsigned short*)(ws + 0);
    float*          c0     = (float*)(ws + 262144);
    float*          b2     = (float*)(ws + 262656);
    unsigned short* w2fr_w = (unsigned short*)(ws + 264704);

    precompute_qa<<<16, 256, 0, stream>>>(se, wts, wq_w, wq_b, wk_w, wk_b, afr_w, c0);
    precompute_w2<<<512, 256, 0, stream>>>(wo_w, wv_w, wv_b, w2fr_w, b2);
    fused_main<<<dim3(64, 16), 512, 0, stream>>>(
        x, (const __bf16*)w2fr_w, (const __bf16*)afr_w, c0, b2, wo_b, out);
}
